// SemanticAttentionTransformerLayer_40630390620222
// MI455X (gfx1250) — matrix operand order repaired and checked by reading, not compiled, not run on hardware
//
#include <hip/hip_runtime.h>

#define DEV __device__ __forceinline__

typedef __attribute__((ext_vector_type(16))) __bf16 bf16x16;
typedef __attribute__((ext_vector_type(8)))  __bf16 bf16x8;
typedef __attribute__((ext_vector_type(8)))  float  f32x8;

static constexpr int Bz = 8, Tz = 1024, Dz = 512, Hz = 8, Sz = 256, FFz = 2048;
static constexpr int AUDz = 768, TXTz = 256, DHz = 64;
static constexpr int BT = Bz * Tz;

DEV f32x8 wmma_bf16(bf16x16 a, bf16x16 b, f32x8 c) {
  return __builtin_amdgcn_wmma_f32_16x16x32_bf16(false, a, false, b, (short)0, c, false, false);
}
DEV bf16x16 cat8(bf16x8 lo, bf16x8 hi) {
  return __builtin_shufflevector(lo, hi, 0,1,2,3,4,5,6,7,8,9,10,11,12,13,14,15);
}
DEV float silu_f(float x) { return x / (1.f + __expf(-x)); }
DEV float gelu_f(float x) { return 0.5f * x * (1.f + erff(x * 0.70710678118f)); }
DEV unsigned pack_bf16(float a, float b) {
  union { __bf16 h[2]; unsigned u; } p;
  p.h[0] = (__bf16)a;
  p.h[1] = (__bf16)b;
  return p.u;
}

// ---------------------------------------------------------------------------
// Generic batched GEMM:  C[m,n] = EPI( A'[m,:] @ W[:,n] + bias[n] ) + rscale*resid[m,n]
//   A' = PREACT(A)  (optionally transposed: A[k,m])
// Block tile 128x128, 8 waves, each wave 32(M)x64(N), K-step 32, bf16 WMMA.
// Register-staged double-buffered LDS (global fetch of tile k+1 overlaps WMMA
// on tile k), packed 32-bit bf16 LDS stores, prefetch of tile k+2.
// Batch: blockIdx.z = zb*ZH + zh, per-operand (b,h) strides.
// ---------------------------------------------------------------------------
template <int PREACT, int EPI, bool TRANSA>
__global__ __launch_bounds__(256) void gemm_k(
    const float* __restrict__ A, long Ab, long Ah,
    const float* __restrict__ W, long Wb, long Wh,
    const float* __restrict__ bias, long Bb, long Bh,
    const float* __restrict__ resid, float rscale,
    float* __restrict__ C, long Cb, long Ch,
    int M, int N, int K, int lda, int ldw, int ldc, int ZH) {
  const int zb = blockIdx.z / ZH, zh = blockIdx.z % ZH;
  A += zb * Ab + zh * Ah;
  W += zb * Wb + zh * Wh;
  if (bias) bias += zb * Bb + zh * Bh;
  C += zb * Cb + zh * Ch;
  if (resid) resid += zb * Cb + zh * Ch;

  const int m0 = blockIdx.y * 128, n0 = blockIdx.x * 128;
  __shared__ __align__(16) __bf16 As[2][128][40];
  __shared__ __align__(16) __bf16 Bs[2][128][40];

  const int tid = threadIdx.x;
  const int lane = tid & 31, hl = lane >> 4, l16 = lane & 15;
  const int wv = tid >> 5, wm = wv >> 1, wn = wv & 1;

  // A staging mapping: thread covers one half-row (16 consecutive k)
  const int amm = tid >> 1, akb = (tid & 1) << 4;

  f32x8 acc[2][4] = {};
  float ra[16], rb[16];

  auto fetchA = [&](int k0) {
#pragma unroll
    for (int i = 0; i < 16; ++i) {
      const int gm = m0 + amm, gk = k0 + akb + i;
      float val = 0.f;
      if (gm < M && gk < K)
        val = TRANSA ? A[(long)gk * lda + gm] : A[(long)gm * lda + gk];
      if (PREACT == 1) val = silu_f(val);
      ra[i] = val;
    }
  };
  auto stageA = [&](int buf) {
#pragma unroll
    for (int i = 0; i < 8; ++i)
      *(unsigned*)&As[buf][amm][akb + 2 * i] = pack_bf16(ra[2 * i], ra[2 * i + 1]);
  };
  // B staging mapping: thread covers 8 (n, k-pair) cells; pairs along k pack
  auto fetchB = [&](int k0) {
#pragma unroll
    for (int i = 0; i < 8; ++i) {
      const int e = (i << 8) + tid;
      const int nn = e & 127, kk = (e >> 7) << 1;
      const int gk = k0 + kk, gn = n0 + nn;
      rb[2 * i]     = (gk < K && gn < N)     ? W[(long)gk * ldw + gn]       : 0.f;
      rb[2 * i + 1] = (gk + 1 < K && gn < N) ? W[(long)(gk + 1) * ldw + gn] : 0.f;
    }
  };
  auto stageB = [&](int buf) {
#pragma unroll
    for (int i = 0; i < 8; ++i) {
      const int e = (i << 8) + tid;
      const int nn = e & 127, kk = (e >> 7) << 1;
      *(unsigned*)&Bs[buf][nn][kk] = pack_bf16(rb[2 * i], rb[2 * i + 1]);
    }
  };

  const int nk = (K + 31) >> 5;
  fetchA(0);
  fetchB(0);
  for (int kt = 0; kt < nk; ++kt) {
    const int cb = kt & 1;
    stageA(cb);
    stageB(cb);
    __syncthreads();
    if (kt + 1 < nk) {
      fetchA((kt + 1) << 5);
      fetchB((kt + 1) << 5);
    }
    if (kt + 2 < nk) {
      const int pk = (kt + 2) << 5;
      if (m0 + amm < M)
        __builtin_prefetch(TRANSA ? &A[(long)(pk)*lda + m0 + amm]
                                  : &A[(long)(m0 + amm) * lda + pk], 0, 1);
      __builtin_prefetch(&W[(long)pk * ldw + n0 + (tid & 127)], 0, 1);
    }

    bf16x16 af[2], bfv[4];
#pragma unroll
    for (int mi = 0; mi < 2; ++mi) {
      const int r = wm * 32 + mi * 16 + l16;
      af[mi] = cat8(*(const bf16x8*)&As[cb][r][8 * hl],
                    *(const bf16x8*)&As[cb][r][16 + 8 * hl]);
    }
#pragma unroll
    for (int ni = 0; ni < 4; ++ni) {
      const int c = wn * 64 + ni * 16 + l16;
      bfv[ni] = cat8(*(const bf16x8*)&Bs[cb][c][8 * hl],
                     *(const bf16x8*)&Bs[cb][c][16 + 8 * hl]);
    }
#pragma unroll
    for (int mi = 0; mi < 2; ++mi)
#pragma unroll
      for (int ni = 0; ni < 4; ++ni)
        acc[mi][ni] = wmma_bf16(af[mi], bfv[ni], acc[mi][ni]);
  }

  // epilogue
#pragma unroll
  for (int mi = 0; mi < 2; ++mi) {
#pragma unroll
    for (int ni = 0; ni < 4; ++ni) {
      const int gc = n0 + wn * 64 + ni * 16 + l16;
      if (gc >= N) continue;
      const float bv = bias ? bias[gc] : 0.f;
#pragma unroll
      for (int j = 0; j < 8; ++j) {
        const int gm = m0 + wm * 32 + mi * 16 + 8 * hl + j;
        if (gm >= M) continue;
        float val = acc[mi][ni][j] + bv;
        if (EPI == 1) val = gelu_f(val);
        if (resid) val += rscale * resid[(long)gm * ldc + gc];
        C[(long)gm * ldc + gc] = val;
      }
    }
  }
}

// ---------------------------------------------------------------------------
// LayerNorm over rows of 512, one wave per row.
// ---------------------------------------------------------------------------
__global__ __launch_bounds__(256) void ln512_k(const float* __restrict__ x,
                                               const float* __restrict__ g,
                                               const float* __restrict__ b,
                                               float* __restrict__ y, int rows) {
  const int w = threadIdx.x >> 5, lane = threadIdx.x & 31;
  const int row = blockIdx.x * 8 + w;
  if (row >= rows) return;
  const float* xr = x + (long)row * 512;
  float v[16], s = 0.f, ss = 0.f;
#pragma unroll
  for (int i = 0; i < 16; ++i) {
    v[i] = xr[lane + i * 32];
    s += v[i];
    ss += v[i] * v[i];
  }
#pragma unroll
  for (int m = 16; m >= 1; m >>= 1) {
    s += __shfl_xor(s, m, 32);
    ss += __shfl_xor(ss, m, 32);
  }
  const float mean = s * (1.f / 512.f);
  const float var = ss * (1.f / 512.f) - mean * mean;
  const float rstd = rsqrtf(var + 1e-5f);
  float* yr = y + (long)row * 512;
#pragma unroll
  for (int i = 0; i < 16; ++i) {
    const int c = lane + i * 32;
    yr[c] = (v[i] - mean) * rstd * g[c] + b[c];
  }
}

// ---------------------------------------------------------------------------
// Flash self-attention: per (b, h, 128-query tile); 8 waves, 16 query rows each.
// dh = 64, key tiles of 64 staged in LDS (K row-major, V transposed), packed
// 32-bit bf16 stores.
// ---------------------------------------------------------------------------
__global__ __launch_bounds__(256) void flash_attn_k(const float* __restrict__ q,
                                                    const float* __restrict__ k,
                                                    const float* __restrict__ v,
                                                    float* __restrict__ o) {
  const int b = blockIdx.z, hh = blockIdx.y, qt = blockIdx.x;
  const int tid = threadIdx.x, w = tid >> 5, lane = tid & 31;
  const int hl = lane >> 4, l16 = lane & 15;

  __shared__ __align__(16) __bf16 Ks[64][72];
  __shared__ __align__(16) __bf16 Vs[64][72];
  __shared__ __align__(16) __bf16 Ps[8][16][72];

  // Q fragments (A-layout) straight from global
  const int qrow = qt * 128 + w * 16 + l16;
  const float* qp = q + ((long)(b * Tz + qrow)) * Dz + hh * DHz;
  bf16x16 qf[2];
#pragma unroll
  for (int ks = 0; ks < 2; ++ks) {
    const int b0 = ks * 32 + 8 * hl, b1 = ks * 32 + 16 + 8 * hl;
    bf16x16 t;
#pragma unroll
    for (int i = 0; i < 8; ++i) {
      t[i] = (__bf16)qp[b0 + i];
      t[8 + i] = (__bf16)qp[b1 + i];
    }
    qf[ks] = t;
  }

  float mrun[8], lrun[8];
  f32x8 oacc[4] = {};
#pragma unroll
  for (int j = 0; j < 8; ++j) { mrun[j] = -1e30f; lrun[j] = 0.f; }

  for (int kb = 0; kb < Tz / 64; ++kb) {
    __syncthreads();
    // K tile: [key][d], packed pairs along d
#pragma unroll
    for (int i = 0; i < 8; ++i) {
      const int e = (i << 8) + tid;
      const int key = e >> 5, d2 = (e & 31) << 1;
      const long g = ((long)(b * Tz + kb * 64 + key)) * Dz + hh * DHz + d2;
      *(unsigned*)&Ks[key][d2] = pack_bf16(k[g], k[g + 1]);
    }
    // V tile transposed: [d][key], packed pairs along key
#pragma unroll
    for (int i = 0; i < 8; ++i) {
      const int e = (i << 8) + tid;
      const int d = e & 63, kp = (e >> 6) << 1;
      const long g = ((long)(b * Tz + kb * 64 + kp)) * Dz + hh * DHz + d;
      *(unsigned*)&Vs[d][kp] = pack_bf16(v[g], v[g + Dz]);
    }
    __syncthreads();

    // scores: 16 queries x 64 keys
    f32x8 s[4] = {};
#pragma unroll
    for (int nt = 0; nt < 4; ++nt) {
      const int key = nt * 16 + l16;
#pragma unroll
      for (int ks = 0; ks < 2; ++ks) {
        bf16x16 bfr = cat8(*(const bf16x8*)&Ks[key][ks * 32 + 8 * hl],
                           *(const bf16x8*)&Ks[key][ks * 32 + 16 + 8 * hl]);
        s[nt] = wmma_bf16(qf[ks], bfr, s[nt]);
      }
      s[nt] = s[nt] * 0.125f;  // 1/sqrt(64)
    }

    // online softmax (row reductions within 16-lane half)
#pragma unroll
    for (int j = 0; j < 8; ++j) {
      float rm = -1e30f;
#pragma unroll
      for (int nt = 0; nt < 4; ++nt) rm = fmaxf(rm, s[nt][j]);
#pragma unroll
      for (int m = 8; m >= 1; m >>= 1) rm = fmaxf(rm, __shfl_xor(rm, m, 16));
      const float mnew = fmaxf(mrun[j], rm);
      const float scl = __expf(mrun[j] - mnew);
      float rs = 0.f;
#pragma unroll
      for (int nt = 0; nt < 4; ++nt) {
        const float e = __expf(s[nt][j] - mnew);
        s[nt][j] = e;
        rs += e;
      }
#pragma unroll
      for (int m = 8; m >= 1; m >>= 1) rs += __shfl_xor(rs, m, 16);
      lrun[j] = lrun[j] * scl + rs;
      mrun[j] = mnew;
#pragma unroll
      for (int dt = 0; dt < 4; ++dt) oacc[dt][j] *= scl;
#pragma unroll
      for (int nt = 0; nt < 4; ++nt)
        Ps[w][8 * hl + j][nt * 16 + l16] = (__bf16)s[nt][j];
    }

    // O += P @ V
#pragma unroll
    for (int ks = 0; ks < 2; ++ks) {
      bf16x16 af = cat8(*(const bf16x8*)&Ps[w][l16][ks * 32 + 8 * hl],
                        *(const bf16x8*)&Ps[w][l16][ks * 32 + 16 + 8 * hl]);
#pragma unroll
      for (int dt = 0; dt < 4; ++dt) {
        const int d = dt * 16 + l16;
        bf16x16 bfr = cat8(*(const bf16x8*)&Vs[d][ks * 32 + 8 * hl],
                           *(const bf16x8*)&Vs[d][ks * 32 + 16 + 8 * hl]);
        oacc[dt] = wmma_bf16(af, bfr, oacc[dt]);
      }
    }
  }

#pragma unroll
  for (int j = 0; j < 8; ++j) {
    const float inv = 1.f / lrun[j];
    const int t = qt * 128 + w * 16 + 8 * hl + j;
    float* op = o + ((long)(b * Tz + t)) * Dz + hh * DHz;
#pragma unroll
    for (int dt = 0; dt < 4; ++dt) op[dt * 16 + l16] = oacc[dt][j] * inv;
  }
}

// ---------------------------------------------------------------------------
// softmax over contiguous rows of 64 (one wave per row) — for cq.
// ---------------------------------------------------------------------------
__global__ __launch_bounds__(256) void softmax64_k(float* __restrict__ x, int rows) {
  const int w = threadIdx.x >> 5, lane = threadIdx.x & 31;
  const int row = blockIdx.x * 8 + w;
  if (row >= rows) return;
  float* p = x + (long)row * 64;
  float a = p[lane], c = p[lane + 32];
  float m = fmaxf(a, c);
#pragma unroll
  for (int ms = 16; ms >= 1; ms >>= 1) m = fmaxf(m, __shfl_xor(m, ms, 32));
  const float ea = __expf(a - m), ec = __expf(c - m);
  float s = ea + ec;
#pragma unroll
  for (int ms = 16; ms >= 1; ms >>= 1) s += __shfl_xor(s, ms, 32);
  const float inv = 1.f / s;
  p[lane] = ea * inv;
  p[lane + 32] = ec * inv;
}

// softmax over token axis: x[b, r, c], r in [0,L), per (b,c)
__global__ void softmax_col_k(float* __restrict__ x, int Bv, int L, int C) {
  const int idx = blockIdx.x * blockDim.x + threadIdx.x;
  if (idx >= Bv * C) return;
  const int b = idx / C, c = idx % C;
  float* p = x + (long)b * L * C + c;
  float m = -1e30f;
  for (int r = 0; r < L; ++r) m = fmaxf(m, p[(long)r * C]);
  float s = 0.f;
  for (int r = 0; r < L; ++r) {
    const float e = __expf(p[(long)r * C] - m);
    p[(long)r * C] = e;
    s += e;
  }
  const float inv = 1.f / s;
  for (int r = 0; r < L; ++r) p[(long)r * C] *= inv;
}

// concat [xw ; broadcast(xf_p)] -> [B, 512, 256]
__global__ void concat_k(const float* __restrict__ xw, const float* __restrict__ xfp,
                         float* __restrict__ out, int total) {
  const int idx = blockIdx.x * blockDim.x + threadIdx.x;
  if (idx >= total) return;
  const int c = idx & 255, r = (idx >> 8) & 511, b = idx >> 17;
  out[idx] = (r < 256) ? xw[((long)b * 256 + r) * 256 + c] : xfp[b * 256 + c];
}

// wsum[b,s,h] = sum_d sk[b,s,h*64+d]
__global__ void wsum_k(const float* __restrict__ sk, float* __restrict__ wsm, int total) {
  const int idx = blockIdx.x * blockDim.x + threadIdx.x;
  if (idx >= total) return;
  const int h = idx & 7, s = (idx >> 3) & 255, b = idx >> 11;
  const float* p = sk + ((long)(b * Sz + s)) * Dz + h * DHz;
  float acc = 0.f;
  for (int d = 0; d < 64; ++d) acc += p[d];
  wsm[idx] = acc;
}

// syv[b,h*64+l] = sum_s wsum[b,s,h] * sv[b,s,h*64+l]
__global__ void syv_k(const float* __restrict__ wsm, const float* __restrict__ sv,
                      float* __restrict__ sy, int total) {
  const int idx = blockIdx.x * blockDim.x + threadIdx.x;
  if (idx >= total) return;
  const int l = idx & 63, h = (idx >> 6) & 7, b = idx >> 9;
  float acc = 0.f;
  for (int s = 0; s < Sz; ++s)
    acc += wsm[((long)b * Sz + s) * Hz + h] * sv[((long)(b * Sz + s)) * Dz + h * DHz + l];
  sy[idx] = acc;
}

// ---------------------------------------------------------------------------
// Host-side launch helper
// ---------------------------------------------------------------------------
struct GemmArgs {
  const float *A, *W, *bias, *resid;
  long Ab, Ah, Wb, Wh, Bb, Bh;
  float rscale;
  float* C;
  long Cb, Ch;
  int M, N, K, lda, ldw, ldc, ZB, ZH;
};

static void launch_gemm(hipStream_t st, const GemmArgs& a, int preact, int epi, bool transa) {
  dim3 grid((a.N + 127) / 128, (a.M + 127) / 128, a.ZB * a.ZH);
  dim3 blk(256);
  if (transa)
    gemm_k<0, 0, true><<<grid, blk, 0, st>>>(a.A, a.Ab, a.Ah, a.W, a.Wb, a.Wh, a.bias, a.Bb,
                                             a.Bh, a.resid, a.rscale, a.C, a.Cb, a.Ch, a.M,
                                             a.N, a.K, a.lda, a.ldw, a.ldc, a.ZH);
  else if (preact == 1)
    gemm_k<1, 0, false><<<grid, blk, 0, st>>>(a.A, a.Ab, a.Ah, a.W, a.Wb, a.Wh, a.bias, a.Bb,
                                              a.Bh, a.resid, a.rscale, a.C, a.Cb, a.Ch, a.M,
                                              a.N, a.K, a.lda, a.ldw, a.ldc, a.ZH);
  else if (epi == 1)
    gemm_k<0, 1, false><<<grid, blk, 0, st>>>(a.A, a.Ab, a.Ah, a.W, a.Wb, a.Wh, a.bias, a.Bb,
                                              a.Bh, a.resid, a.rscale, a.C, a.Cb, a.Ch, a.M,
                                              a.N, a.K, a.lda, a.ldw, a.ldc, a.ZH);
  else
    gemm_k<0, 0, false><<<grid, blk, 0, st>>>(a.A, a.Ab, a.Ah, a.W, a.Wb, a.Wh, a.bias, a.Bb,
                                              a.Bh, a.resid, a.rscale, a.C, a.Cb, a.Ch, a.M,
                                              a.N, a.K, a.lda, a.ldw, a.ldc, a.ZH);
}

static GemmArgs simple_gemm(const float* A, const float* W, const float* bias,
                            const float* resid, float rscale, float* C, int M, int N, int K) {
  GemmArgs g{};
  g.A = A; g.W = W; g.bias = bias; g.resid = resid; g.rscale = rscale; g.C = C;
  g.M = M; g.N = N; g.K = K; g.lda = K; g.ldw = N; g.ldc = N;
  g.ZB = 1; g.ZH = 1;
  return g;
}

extern "C" void kernel_launch(void* const* d_in, const int* in_sizes, int n_in,
                              void* d_out, int out_size, void* d_ws, size_t ws_size,
                              hipStream_t stream) {
  (void)in_sizes; (void)n_in; (void)out_size; (void)ws_size;
  const float* x      = (const float*)d_in[0];
  const float* xf     = (const float*)d_in[1];
  const float* xw     = (const float*)d_in[2];
  const float* xs     = (const float*)d_in[3];
  const float* fp_w   = (const float*)d_in[4];
  const float* fp_b   = (const float*)d_in[5];
  const float* sa_ng  = (const float*)d_in[6];
  const float* sa_nb  = (const float*)d_in[7];
  const float* sa_qw  = (const float*)d_in[8];
  const float* sa_qb  = (const float*)d_in[9];
  const float* sa_kw  = (const float*)d_in[10];
  const float* sa_kb  = (const float*)d_in[11];
  const float* sa_vw  = (const float*)d_in[12];
  const float* sa_vb  = (const float*)d_in[13];
  const float* sa_ow  = (const float*)d_in[14];
  const float* sa_ob  = (const float*)d_in[15];
  const float* ca_ng  = (const float*)d_in[16];
  const float* ca_nb  = (const float*)d_in[17];
  const float* ca_tng = (const float*)d_in[18];
  const float* ca_tnb = (const float*)d_in[19];
  const float* ca_sng = (const float*)d_in[20];
  const float* ca_snb = (const float*)d_in[21];
  const float* ca_qw  = (const float*)d_in[22];
  const float* ca_qb  = (const float*)d_in[23];
  const float* ca_kw  = (const float*)d_in[24];
  const float* ca_kb  = (const float*)d_in[25];
  const float* ca_vw  = (const float*)d_in[26];
  const float* ca_vb  = (const float*)d_in[27];
  const float* ca_apw = (const float*)d_in[28];
  const float* ca_apb = (const float*)d_in[29];
  const float* ca_atw = (const float*)d_in[30];
  const float* ca_atb = (const float*)d_in[31];
  const float* ca_ow  = (const float*)d_in[32];
  const float* ca_ob  = (const float*)d_in[33];
  const float* an_g   = (const float*)d_in[34];
  const float* an_b   = (const float*)d_in[35];
  const float* ffn_w1 = (const float*)d_in[36];
  const float* ffn_b1 = (const float*)d_in[37];
  const float* ffn_w2 = (const float*)d_in[38];
  const float* ffn_b2 = (const float*)d_in[39];
  const float* ffn_ng = (const float*)d_in[40];
  const float* ffn_nb = (const float*)d_in[41];
  const float* ffn_ow = (const float*)d_in[42];
  const float* ffn_ob = (const float*)d_in[43];
  float* out = (float*)d_out;

  float* ws = (float*)d_ws;
  size_t off = 0;
  auto alloc = [&](size_t n) { float* p = ws + off; off += n; return p; };
  const size_t SZ = (size_t)BT * Dz;
  float* A0 = alloc(SZ);
  float* A1 = alloc(SZ);
  float* A2 = alloc(SZ);
  float* A3 = alloc(SZ);
  float* A4 = alloc(SZ);
  float* Hb = alloc((size_t)BT * FFz);
  float* xfp  = alloc((size_t)Bz * TXTz);
  float* xcin = alloc((size_t)Bz * 512 * TXTz);
  float* xwf  = alloc((size_t)Bz * 512 * Dz);
  float* tn   = alloc((size_t)Bz * 512 * Dz);
  float* sn   = alloc((size_t)Bz * Sz * Dz);
  float* ck   = alloc((size_t)Bz * 512 * Dz);
  float* cv   = alloc((size_t)Bz * 512 * Dz);
  float* sk   = alloc((size_t)Bz * Sz * Dz);
  float* sv   = alloc((size_t)Bz * Sz * Dz);
  float* attb = alloc((size_t)Bz * Hz * DHz * DHz);
  float* wsm  = alloc((size_t)Bz * Sz * Hz);
  float* syv  = alloc((size_t)Bz * Hz * DHz);

  const dim3 b256(256);

  // 1. xp = x @ fp_w + fp_b + x                          -> A0
  launch_gemm(stream, simple_gemm(x, fp_w, fp_b, x, 1.f, A0, BT, Dz, Dz), 0, 0, false);
  // 2. xn = LN(xp; sa_ng, sa_nb)                         -> A1
  ln512_k<<<dim3(BT / 8), b256, 0, stream>>>(A0, sa_ng, sa_nb, A1, BT);
  // 3. q, k, v                                           -> A2, A3, A4
  launch_gemm(stream, simple_gemm(A1, sa_qw, sa_qb, nullptr, 0.f, A2, BT, Dz, Dz), 0, 0, false);
  launch_gemm(stream, simple_gemm(A1, sa_kw, sa_kb, nullptr, 0.f, A3, BT, Dz, Dz), 0, 0, false);
  launch_gemm(stream, simple_gemm(A1, sa_vw, sa_vb, nullptr, 0.f, A4, BT, Dz, Dz), 0, 0, false);
  // 4. ao = flash-attn(q,k,v)                            -> A1
  flash_attn_k<<<dim3(Tz / 128, Hz, Bz), b256, 0, stream>>>(A2, A3, A4, A1);
  // 5. t1 = silu(ao) @ sa_ow + sa_ob + 2*xp              -> A2
  launch_gemm(stream, simple_gemm(A1, sa_ow, sa_ob, A0, 2.f, A2, BT, Dz, Dz), 1, 0, false);
  // 6. y1 = LN(t1; an)                                   -> A3
  ln512_k<<<dim3(BT / 8), b256, 0, stream>>>(A2, an_g, an_b, A3, BT);
  // 7. xf_p = xf @ ca_apw + ca_apb                       -> xfp
  launch_gemm(stream, simple_gemm(xf, ca_apw, ca_apb, nullptr, 0.f, xfp, Bz, TXTz, AUDz), 0, 0, false);
  // 8. concat [xw ; xf_p]                                -> xcin
  {
    const int total = Bz * 512 * TXTz;
    concat_k<<<dim3((total + 255) / 256), b256, 0, stream>>>(xw, xfp, xcin, total);
  }
  // 9. xwf = xcin @ ca_atw + ca_atb                      -> xwf
  launch_gemm(stream, simple_gemm(xcin, ca_atw, ca_atb, nullptr, 0.f, xwf, Bz * 512, Dz, TXTz), 0, 0, false);
  // 10. tn = LN(xwf; ca_tn)                              -> tn
  ln512_k<<<dim3(Bz * 512 / 8), b256, 0, stream>>>(xwf, ca_tng, ca_tnb, tn, Bz * 512);
  // 11. sn = LN(xs; ca_sn)                               -> sn
  ln512_k<<<dim3(Bz * Sz / 8), b256, 0, stream>>>(xs, ca_sng, ca_snb, sn, Bz * Sz);
  // 12. yn = LN(y1; ca_n)                                -> A0
  ln512_k<<<dim3(BT / 8), b256, 0, stream>>>(A3, ca_ng, ca_nb, A0, BT);
  // 13. cq = softmax64(yn @ ca_qw + ca_qb)               -> A4
  launch_gemm(stream, simple_gemm(A0, ca_qw, ca_qb, nullptr, 0.f, A4, BT, Dz, Dz), 0, 0, false);
  softmax64_k<<<dim3(BT * Hz / 8), b256, 0, stream>>>(A4, BT * Hz);
  // 14. ck = softmax_tokens(tn @ ca_kw + ca_kb)          -> ck
  launch_gemm(stream, simple_gemm(tn, ca_kw, ca_kb, nullptr, 0.f, ck, Bz * 512, Dz, Dz), 0, 0, false);
  softmax_col_k<<<dim3((Bz * Dz + 255) / 256), b256, 0, stream>>>(ck, Bz, 512, Dz);
  // 15. sk = softmax_tokens(sn @ ca_kw + ca_kb)          -> sk
  launch_gemm(stream, simple_gemm(sn, ca_kw, ca_kb, nullptr, 0.f, sk, Bz * Sz, Dz, Dz), 0, 0, false);
  softmax_col_k<<<dim3((Bz * Dz + 255) / 256), b256, 0, stream>>>(sk, Bz, Sz, Dz);
  // 16. cv, sv
  launch_gemm(stream, simple_gemm(tn, ca_vw, ca_vb, nullptr, 0.f, cv, Bz * 512, Dz, Dz), 0, 0, false);
  launch_gemm(stream, simple_gemm(sn, ca_vw, ca_vb, nullptr, 0.f, sv, Bz * Sz, Dz, Dz), 0, 0, false);
  // 17. att[b,h] = ck_bh^T @ cv_bh   (64x64, K=512)      -> attb
  {
    GemmArgs g{};
    g.A = ck; g.Ab = (long)512 * Dz; g.Ah = DHz; g.lda = Dz;
    g.W = cv; g.Wb = (long)512 * Dz; g.Wh = DHz; g.ldw = Dz;
    g.bias = nullptr; g.resid = nullptr; g.rscale = 0.f;
    g.C = attb; g.Cb = (long)Hz * DHz * DHz; g.Ch = (long)DHz * DHz; g.ldc = DHz;
    g.M = DHz; g.N = DHz; g.K = 512; g.ZB = Bz; g.ZH = Hz;
    launch_gemm(stream, g, 0, 0, true);
  }
  // 18. wsum, syv
  wsum_k<<<dim3((Bz * Sz * Hz + 255) / 256), b256, 0, stream>>>(sk, wsm, Bz * Sz * Hz);
  syv_k<<<dim3((Bz * Hz * DHz + 255) / 256), b256, 0, stream>>>(wsm, sv, syv, Bz * Hz * DHz);
  // 19. cy[b,:,h] = cq_bh @ att_bh + syv_bh (col bias)   -> A1
  {
    GemmArgs g{};
    g.A = A4; g.Ab = (long)Tz * Dz; g.Ah = DHz; g.lda = Dz;
    g.W = attb; g.Wb = (long)Hz * DHz * DHz; g.Wh = (long)DHz * DHz; g.ldw = DHz;
    g.bias = syv; g.Bb = (long)Hz * DHz; g.Bh = DHz;
    g.resid = nullptr; g.rscale = 0.f;
    g.C = A1; g.Cb = (long)Tz * Dz; g.Ch = DHz; g.ldc = Dz;
    g.M = Tz; g.N = DHz; g.K = DHz; g.ZB = Bz; g.ZH = Hz;
    launch_gemm(stream, g, 0, 0, false);
  }
  // 20. t2 = silu(cy) @ ca_ow + ca_ob + 2*y1             -> A2
  launch_gemm(stream, simple_gemm(A1, ca_ow, ca_ob, A3, 2.f, A2, BT, Dz, Dz), 1, 0, false);
  // 21. y2 = LN(t2; an)                                  -> A0
  ln512_k<<<dim3(BT / 8), b256, 0, stream>>>(A2, an_g, an_b, A0, BT);
  // 22. h1 = gelu(y2 @ w1 + b1)                          -> Hb
  launch_gemm(stream, simple_gemm(A0, ffn_w1, ffn_b1, nullptr, 0.f, Hb, BT, FFz, Dz), 0, 1, false);
  // 23. h2l = h1 @ w2 + b2                               -> A1
  launch_gemm(stream, simple_gemm(Hb, ffn_w2, ffn_b2, nullptr, 0.f, A1, BT, Dz, FFz), 0, 0, false);
  // 24. h2 = LN(h2l; ffn_n)                              -> A4
  ln512_k<<<dim3(BT / 8), b256, 0, stream>>>(A1, ffn_ng, ffn_nb, A4, BT);
  // 25. t3 = silu(h2) @ ffn_ow + ffn_ob + 2*y2           -> A1
  launch_gemm(stream, simple_gemm(A4, ffn_ow, ffn_ob, A0, 2.f, A1, BT, Dz, Dz), 1, 0, false);
  // 26. y3 = LN(t3; an)                                  -> out
  ln512_k<<<dim3(BT / 8), b256, 0, stream>>>(A1, an_g, an_b, out, BT);
}
